// SelectorNN_73040213835925
// MI455X (gfx1250) — matrix operand order repaired and checked by reading, not compiled, not run on hardware
//
#include <hip/hip_runtime.h>

// Problem constants (match reference)
#define VOCAB 50000
#define D     300
#define DP    320     // K padded to multiple of 32 for bf16 WMMA
#define B_    64
#define LC    32
#define T_    256
#define LT    64
#define NSEL  5
#define TCHUNK 4      // sentences per workgroup: amortize claim staging

typedef __attribute__((ext_vector_type(16))) __bf16 v16bf;
typedef __attribute__((ext_vector_type(8)))  float  v8f;
typedef __attribute__((ext_vector_type(4)))  int    v4i;

// CDNA5 async global->LDS copy (ASYNCcnt-tracked), if toolchain exposes it.
#if __has_builtin(__builtin_amdgcn_global_load_async_to_lds_b128) && \
    __has_builtin(__builtin_amdgcn_s_wait_asynccnt)
#define USE_ASYNC_LDS 1
#else
#define USE_ASYNC_LDS 0
#endif

#if USE_ASYNC_LDS
// Prototype (from hipcc diagnostic): param0 = v4i addrspace(1)*, param1 = LDS ptr,
// then two immediates (offset, cpol).
typedef __attribute__((address_space(1))) v4i* gptr_v4i;
typedef __attribute__((address_space(3))) v4i* lptr_v4i;

__device__ __forceinline__ void async_cp16(const void* g, void* l) {
  __builtin_amdgcn_global_load_async_to_lds_b128((gptr_v4i)g, (lptr_v4i)l, 0, 0);
}
#endif

// ---------------------------------------------------------------------------
// Kernel 0: convert f32 embedding table -> bf16, K-padded to DP (zeros).
// 32 MB result lives in workspace; fits MI455X 192MB L2 -> gathers stay on-chip.
// ---------------------------------------------------------------------------
__global__ void __launch_bounds__(256)
cvt_table(const float* __restrict__ emb, unsigned short* __restrict__ tab) {
  int idx = blockIdx.x * 256 + threadIdx.x;
  if (idx >= VOCAB * DP) return;
  int row = idx / DP, k = idx - row * DP;
  float v = (k < D) ? emb[(size_t)row * D + k] : 0.0f;
  __bf16 h = (__bf16)v;
  tab[idx] = __builtin_bit_cast(unsigned short, h);
}

// ---------------------------------------------------------------------------
// Kernel 1: per-(b,t) 32x64 similarity tile via v_wmma_f32_16x16x32_bf16,
// then softmax(axis=LT) -> max(axis=LC) -> sum  => target_scores[b][t].
// Block = 256 threads = 8 wave32; wave w owns output subtile (w>>2, w&3).
// Each workgroup handles TCHUNK sentences, reusing the staged claim tile.
// ---------------------------------------------------------------------------
__global__ void __launch_bounds__(256)
score_tiles(const int* __restrict__ claim, const int* __restrict__ targets,
            const unsigned short* __restrict__ tab, float* __restrict__ scores) {
  const int t0  = blockIdx.x * TCHUNK;
  const int b   = blockIdx.y;
  const int tid = threadIdx.x;

  __shared__ __align__(16) __bf16 sA[LC * DP];   // 20 KB claim rows (persistent)
  __shared__ __align__(16) __bf16 sB[LT * DP];   // 40 KB target rows (re-used for U)

  const int wave = tid >> 5;
  const int lane = tid & 31;
  const int half = lane >> 4;    // which 16-lane half
  const int ln16 = lane & 15;
  const int mi   = wave >> 2;    // 0..1 (M subtile)
  const int ni   = wave & 3;     // 0..3 (N subtile)

  const __bf16* arow = sA + (size_t)(mi * 16 + ln16) * DP;  // claim row m
  const __bf16* brow = sB + (size_t)(ni * 16 + ln16) * DP;  // target row n (B col)

  // ---- stage claim rows once (row = 640 B = 40 x 16 B chunks; 1280 total
  //      -> uniform 5 iterations across 256 lanes, no divergence)
  for (int i = tid; i < LC * 40; i += 256) {
    int row = i / 40, ch = i - row * 40;
    int id  = claim[b * LC + row];
    const char* src = (const char*)(tab + (size_t)id * DP) + ch * 16;
    char*       dst = (char*)sA + (size_t)row * (DP * 2) + ch * 16;
#if USE_ASYNC_LDS
    async_cp16(src, dst);
#else
    *(uint4*)dst = *(const uint4*)src;
#endif
  }

  for (int tt = 0; tt < TCHUNK; ++tt) {
    const int t = t0 + tt;

    // ---- stage 64 target rows for this sentence (2560 chunks -> 10 iters)
    const int* tix = targets + ((size_t)b * T_ + t) * LT;
    for (int i = tid; i < LT * 40; i += 256) {
      int row = i / 40, ch = i - row * 40;
      int id  = tix[row];
      const char* src = (const char*)(tab + (size_t)id * DP) + ch * 16;
      char*       dst = (char*)sB + (size_t)row * (DP * 2) + ch * 16;
#if USE_ASYNC_LDS
      async_cp16(src, dst);
#else
      *(uint4*)dst = *(const uint4*)src;
#endif
    }
#if USE_ASYNC_LDS
    __builtin_amdgcn_s_wait_asynccnt(0);   // own async copies landed in LDS
#endif
    __syncthreads();                       // everyone's data visible

    // ---- WMMA: 2x4 grid of 16x16 subtiles, K = DP in steps of 32
    v8f acc = {};
    #pragma unroll
    for (int kk = 0; kk < DP / 32; ++kk) {
      // A 16x32 bf16 frag: lane holds K = 8*half+[0,8) then 16+8*half+[0,8)
      const __bf16* pa0 = arow + kk * 32 + 8 * half;
      const __bf16* pa1 = pa0 + 16;
      // B 32x16 bf16 frag: lane holds column n, K = 8*half+[0,8) then 16+8*half+[0,8)
      const __bf16* pb  = brow + kk * 32 + 8 * half;
      v16bf af, bf;
      #pragma unroll
      for (int i = 0; i < 8; ++i) { af[i] = pa0[i]; af[i + 8] = pa1[i]; }
      #pragma unroll
      for (int i = 0; i < 8; ++i) { bf[i] = pb[i]; bf[i + 8] = pb[i + 16]; }
      acc = __builtin_amdgcn_wmma_f32_16x16x32_bf16(false, af, false, bf,
                                                    (short)0, acc, false, false);
    }
    __syncthreads();   // all waves done reading sB before aliasing it with U

    float* sU     = (float*)sB;          // 32x64 f32 tile (8 KB) aliases sB
    float* rowmax = sU + LC * LT;
    float* rowinv = rowmax + LC;
    float* colsc  = rowinv + LC;

    // C/D layout: VGPR v of lane -> row mi*16 + 8*half + v, col ni*16 + ln16
    #pragma unroll
    for (int v = 0; v < 8; ++v)
      sU[(mi * 16 + 8 * half + v) * LT + (ni * 16 + ln16)] = acc[v];
    __syncthreads();

    // per claim-row softmax stats (max + 1/sum_exp over LT)
    if (tid < LC) {
      const float* u = sU + tid * LT;
      float m = -1e30f;
      for (int l = 0; l < LT; ++l) m = fmaxf(m, u[l]);
      float s = 0.f;
      for (int l = 0; l < LT; ++l) s += __expf(u[l] - m);
      rowmax[tid] = m;
      rowinv[tid] = 1.0f / s;
    }
    __syncthreads();

    // per target-token: max over claim rows of softmax prob
    if (tid < LT) {
      float cm = 0.f;
      for (int c = 0; c < LC; ++c)
        cm = fmaxf(cm, __expf(sU[c * LT + tid] - rowmax[c]) * rowinv[c]);
      colsc[tid] = cm;
    }
    __syncthreads();

    if (tid == 0) {
      float s = 0.f;
      for (int l = 0; l < LT; ++l) s += colsc[l];
      scores[b * T_ + t] = s;
    }
    __syncthreads();   // sU/stats reads done before next sentence restages sB
  }
}

// ---------------------------------------------------------------------------
// Kernel 2: top-5 per batch (repeated argmax; low index wins ties like top_k)
// ---------------------------------------------------------------------------
__global__ void __launch_bounds__(256)
topk5(const float* __restrict__ scores, int* __restrict__ topk) {
  const int b = blockIdx.x, tid = threadIdx.x;
  __shared__ float sv[256];
  __shared__ int   si[256];
  float v = scores[b * T_ + tid];
  for (int j = 0; j < NSEL; ++j) {
    sv[tid] = v; si[tid] = tid;
    __syncthreads();
    for (int s = 128; s > 0; s >>= 1) {
      if (tid < s) {
        float ov = sv[tid + s]; int oi = si[tid + s];
        if (ov > sv[tid] || (ov == sv[tid] && oi < si[tid])) { sv[tid] = ov; si[tid] = oi; }
      }
      __syncthreads();
    }
    int win = si[0];
    if (tid == 0) topk[b * NSEL + j] = win;
    if (tid == win) v = -1e30f;
    __syncthreads();
  }
}

// ---------------------------------------------------------------------------
// Kernel 3: recompute winning 32x64 tiles in full f32 (0.39 GFLOP total),
// L2-normalize each claim row over LT, write output [B,5,LC,LT].
// ---------------------------------------------------------------------------
__global__ void __launch_bounds__(256)
emit_top(const int* __restrict__ claim, const int* __restrict__ targets,
         const float* __restrict__ emb, const int* __restrict__ topk,
         float* __restrict__ out) {
  const int b   = blockIdx.x / NSEL;
  const int j   = blockIdx.x - b * NSEL;
  const int t   = topk[b * NSEL + j];
  const int tid = threadIdx.x;

  __shared__ __align__(16) float sC[LC * D];   // 38.4 KB claim rows (f32)
  __shared__ float sU[LC * LT];                // 8 KB
  __shared__ float sInv[LC];

  for (int i = tid; i < LC * D; i += 256) {
    int row = i / D, k = i - row * D;
    sC[i] = emb[(size_t)claim[b * LC + row] * D + k];
  }
  __syncthreads();

  const int* tix = targets + ((size_t)b * T_ + t) * LT;
  for (int e = tid; e < LC * LT; e += 256) {
    int c = e >> 6, l = e & 63;
    const float* trow = emb + (size_t)tix[l] * D;
    const float* crow = sC + c * D;
    float s = 0.f;
    for (int k = 0; k < D; ++k) s = fmaf(crow[k], trow[k], s);
    sU[e] = s;
  }
  __syncthreads();

  if (tid < LC) {
    const float* u = sU + tid * LT;
    float s = 0.f;
    for (int l = 0; l < LT; ++l) s = fmaf(u[l], u[l], s);
    sInv[tid] = 1.0f / sqrtf(s);
  }
  __syncthreads();

  float* o = out + (size_t)(b * NSEL + j) * (LC * LT);
  for (int e = tid; e < LC * LT; e += 256) o[e] = sU[e] * sInv[e >> 6];
}

// ---------------------------------------------------------------------------
// Launcher. Workspace layout: [bf16 table 32,000,000 B][scores 65,536 B][topk 1,280 B]
// ---------------------------------------------------------------------------
extern "C" void kernel_launch(void* const* d_in, const int* in_sizes, int n_in,
                              void* d_out, int out_size, void* d_ws, size_t ws_size,
                              hipStream_t stream) {
  (void)in_sizes; (void)n_in; (void)out_size; (void)ws_size;
  const int*   claim   = (const int*)d_in[0];
  const int*   targets = (const int*)d_in[1];
  const float* emb     = (const float*)d_in[2];
  // d_in[3] is n (==5), fixed at compile time.

  const size_t TAB_BYTES = (size_t)VOCAB * DP * 2;   // 32,000,000
  unsigned short* tab    = (unsigned short*)d_ws;
  float*          scores = (float*)((char*)d_ws + TAB_BYTES);
  int*            topk   = (int*)((char*)d_ws + TAB_BYTES + (size_t)B_ * T_ * sizeof(float));

  cvt_table  <<<(VOCAB * DP + 255) / 256, 256, 0, stream>>>(emb, tab);
  score_tiles<<<dim3(T_ / TCHUNK, B_), 256, 0, stream>>>(claim, targets, tab, scores);
  topk5      <<<B_, 256, 0, stream>>>(scores, topk);
  emit_top   <<<B_ * NSEL, 256, 0, stream>>>(claim, targets, emb, topk, (float*)d_out);
}
